// RransformerBlock_41214506172935
// MI455X (gfx1250) — compile-verified, matrix operand order checked by reading, not run on hardware
//
#include <hip/hip_runtime.h>
#include <hip/hip_bf16.h>
#include <math.h>

// ---------------------------------------------------------------------------
// CDNA5 (gfx1250) implementation of the Rransformer block.
// All 1x1 convs run on v_wmma_f32_16x16x32_bf16. Activations are produced
// pre-swizzled in the WMMA B-fragment layout so GEMM lanes do pure b128 loads.
// ---------------------------------------------------------------------------

typedef __bf16 bf16_t;
typedef __attribute__((ext_vector_type(16))) __bf16 v16bf;
typedef __attribute__((ext_vector_type(8)))  __bf16 v8bf;
typedef __attribute__((ext_vector_type(8)))  float  v8f;

#define DIMC   192
#define HEADS  6
#define HIDDEN 384
#define HH     128
#define WW     128
#define PIX    (HH * WW)          // 16384 pixels per batch
#define NB     2                  // batch

__device__ inline bf16_t f2bf(float f) { return (bf16_t)f; }

// ---------------------------------------------------------------------------
// Fragment loader. Storage layout for a 16-row x 32-col (row=M or N, col=K)
// fragment: row-major, 32 bf16 per row (64B). Per the CDNA5 ISA 16-bit A/B
// layout: lanes 0-15 need kk {0..7, 16..23}, lanes 16-31 need {8..15, 24..31},
// i.e. two contiguous 16-byte chunks per lane.
// ---------------------------------------------------------------------------
__device__ inline v16bf load_frag(const bf16_t* __restrict__ base, int lane) {
  int m  = lane & 15;
  int hi = (lane >> 4) & 1;
  const bf16_t* row = base + (size_t)m * 32 + hi * 8;
  v8bf lo = *(const v8bf*)(row);
  v8bf up = *(const v8bf*)(row + 16);
  v16bf r;
#pragma unroll
  for (int i = 0; i < 8; ++i) { r[i] = lo[i]; r[i + 8] = up[i]; }
  return r;
}

// ---------------------------------------------------------------------------
// Generic WMMA GEMM:  out[b, o, p] = sum_k Wf[o,k] * Bact[b,k,p]  (+ residual)
// Wf  : bf16 fragments, layout ((ot*nkt+kt)*16 + m)*32 + kk
// Bact: bf16 fragments, layout ((b*nkt+kt)*P + p)*32 + kk
// grid = (P/128, O/16, NB), block = 32 (one wave per 16oc x 128px tile).
// All 8 B-fragments of a K-step are loaded before the 8 WMMAs so the
// compiler can overlap matrix issue with outstanding loads (partial
// s_wait_loadcnt) instead of draining to zero before each WMMA.
// ---------------------------------------------------------------------------
__global__ void gemm_wmma(const bf16_t* __restrict__ Wf,
                          const bf16_t* __restrict__ Bact,
                          float* __restrict__ out,
                          const float* __restrict__ res,
                          int O, int Kp, int P) {
  const int lane = threadIdx.x;
  const int ot   = blockIdx.y;
  const int bb   = blockIdx.z;
  const int nkt  = Kp >> 5;
  const int p0   = blockIdx.x * 128;

  v8f acc[8];
#pragma unroll
  for (int t = 0; t < 8; ++t) acc[t] = (v8f){0.f,0.f,0.f,0.f,0.f,0.f,0.f,0.f};

  for (int kt = 0; kt < nkt; ++kt) {
    const bf16_t* abase  = Wf + ((size_t)(ot * nkt + kt)) * (16 * 32);
    const bf16_t* bplane = Bact + (((size_t)bb * nkt + kt) * (size_t)P) * 32;
    if (kt + 1 < nkt) {
      const bf16_t* nxt = Bact + (((size_t)bb * nkt + kt + 1) * (size_t)P + p0) * 32;
      __builtin_prefetch(nxt, 0, 1);   // global_prefetch_b8 -> L2 for next K-plane
    }
    const v16bf a = load_frag(abase, lane);
    v16bf bf[8];
#pragma unroll
    for (int t = 0; t < 8; ++t)
      bf[t] = load_frag(bplane + (size_t)(p0 + t * 16) * 32, lane);
#pragma unroll
    for (int t = 0; t < 8; ++t)
      acc[t] = __builtin_amdgcn_wmma_f32_16x16x32_bf16(
          false, a, false, bf[t], (short)0, acc[t], false, false);
  }

  const int m_lo = lane & 15;
  const int hi8  = (lane >> 4) ? 8 : 0;
#pragma unroll
  for (int t = 0; t < 8; ++t) {
    int p = p0 + t * 16 + m_lo;
#pragma unroll
    for (int r = 0; r < 8; ++r) {
      int oc = ot * 16 + r + hi8;
      size_t idx = ((size_t)bb * O + oc) * (size_t)P + p;
      float v = acc[t][r];
      if (res) v += res[idx];
      out[idx] = v;
    }
  }
}

// ---------------------------------------------------------------------------
// Weight prep: f32 [O,K] row-major -> bf16 fragment layout, K padded to Kp.
// ---------------------------------------------------------------------------
__global__ void wprep(const float* __restrict__ W, bf16_t* __restrict__ out,
                      int O, int K, int Kp) {
  int id = blockIdx.x * blockDim.x + threadIdx.x;
  int total = O * Kp;
  if (id >= total) return;
  int kk   = id & 31;
  int rest = id >> 5;
  int m    = rest & 15;
  int frag = rest >> 4;
  int nkt  = Kp >> 5;
  int kt   = frag % nkt, ot = frag / nkt;
  int o = ot * 16 + m, k = kt * 32 + kk;
  out[id] = f2bf((k < K) ? W[(size_t)o * K + k] : 0.f);
}

// ---------------------------------------------------------------------------
// Small projections of k_v: kvr[b,144], kvi[b,48], kvf[b,192]
// ---------------------------------------------------------------------------
__global__ void kvproj(const float* __restrict__ k_v,
                       const float* __restrict__ w_kR,
                       const float* __restrict__ w_kI,
                       const float* __restrict__ w_ffk,
                       float* kvr, float* kvi, float* kvf) {
  int id = blockIdx.x * blockDim.x + threadIdx.x;
  if (id < 288) {
    int bb = id / 144, o = id % 144;
    float s = 0;
    for (int k = 0; k < 192; ++k) s += k_v[bb * 256 + k] * w_kR[o * 192 + k];
    kvr[id] = s;
  } else if (id < 384) {
    int t = id - 288, bb = t / 48, o = t % 48;
    float s = 0;
    for (int k = 0; k < 64; ++k) s += k_v[bb * 256 + 192 + k] * w_kI[o * 64 + k];
    kvi[t] = s;
  } else if (id < 768) {
    int t = id - 384, bb = t / 192, o = t % 192;
    float s = 0;
    for (int k = 0; k < 256; ++k) s += k_v[bb * 256 + k] * w_ffk[o * 256 + k];
    kvf[t] = s;
  }
}

// ---------------------------------------------------------------------------
// LN1 over channels + modulate, write swizzled bf16 x_r (144->160 ch padded)
// and x_i (48->64 ch padded).  grid=(P/256, NB), block=256
// ---------------------------------------------------------------------------
__global__ void ln1_mod(const float* __restrict__ x,
                        const float* __restrict__ w, const float* __restrict__ b,
                        const float* __restrict__ kvr, const float* __restrict__ kvi,
                        bf16_t* __restrict__ M1r, bf16_t* __restrict__ M1i, int P) {
  int p  = blockIdx.x * blockDim.x + threadIdx.x;
  int bb = blockIdx.y;
  if (p >= P) return;
  const float* xp = x + (size_t)bb * DIMC * P + p;
  float s = 0.f, s2 = 0.f;
  for (int c = 0; c < DIMC; ++c) { float v = xp[(size_t)c * P]; s += v; s2 += v * v; }
  float mu  = s * (1.f / DIMC);
  float var = s2 * (1.f / DIMC) - mu * mu;
  float rstd = rsqrtf(var + 1e-5f);
  for (int c = 0; c < DIMC; ++c) {
    float xn = (xp[(size_t)c * P] - mu) * rstd * w[c] + b[c];
    if (c < 144) {
      float m = xn * kvr[bb * 144 + c] + xn;
      M1r[(((size_t)bb * 5 + (c >> 5)) * P + p) * 32 + (c & 31)] = f2bf(m);
    } else {
      int ci = c - 144;
      float m = xn * kvi[bb * 48 + ci] + xn;
      M1i[(((size_t)bb * 2 + (ci >> 5)) * P + p) * 32 + (ci & 31)] = f2bf(m);
    }
  }
  for (int kk = 16; kk < 32; ++kk) {   // zero K-padding lanes
    M1r[(((size_t)bb * 5 + 4) * P + p) * 32 + kk] = f2bf(0.f);
    M1i[(((size_t)bb * 2 + 1) * P + p) * 32 + kk] = f2bf(0.f);
  }
}

// LN2 + modulate by kvf, all 192 channels -> swizzled bf16 (6 tiles)
__global__ void ln2_mod(const float* __restrict__ x,
                        const float* __restrict__ w, const float* __restrict__ b,
                        const float* __restrict__ kvf,
                        bf16_t* __restrict__ M2, int P) {
  int p  = blockIdx.x * blockDim.x + threadIdx.x;
  int bb = blockIdx.y;
  if (p >= P) return;
  const float* xp = x + (size_t)bb * DIMC * P + p;
  float s = 0.f, s2 = 0.f;
  for (int c = 0; c < DIMC; ++c) { float v = xp[(size_t)c * P]; s += v; s2 += v * v; }
  float mu  = s * (1.f / DIMC);
  float var = s2 * (1.f / DIMC) - mu * mu;
  float rstd = rsqrtf(var + 1e-5f);
  for (int c = 0; c < DIMC; ++c) {
    float xn = (xp[(size_t)c * P] - mu) * rstd * w[c] + b[c];
    float m  = xn * kvf[bb * DIMC + c] + xn;
    M2[(((size_t)bb * 6 + (c >> 5)) * P + p) * 32 + (c & 31)] = f2bf(m);
  }
}

// ---------------------------------------------------------------------------
// Depthwise 3x3, planar f32 -> planar f32. grid=(P/256, C, NB)
// ---------------------------------------------------------------------------
__global__ void dwconv3(const float* __restrict__ in, const float* __restrict__ wt,
                        float* __restrict__ out, int C, int P) {
  int p  = blockIdx.x * blockDim.x + threadIdx.x;
  int c  = blockIdx.y;
  int bb = blockIdx.z;
  if (p >= P) return;
  int py = p / WW, px = p % WW;
  const float* ip = in + ((size_t)bb * C + c) * P;
  const float* w9 = wt + c * 9;
  float acc = 0.f;
#pragma unroll
  for (int dy = -1; dy <= 1; ++dy)
#pragma unroll
    for (int dx = -1; dx <= 1; ++dx) {
      int yy = py + dy, xx = px + dx;
      if (yy >= 0 && yy < HH && xx >= 0 && xx < WW)
        acc += w9[(dy + 1) * 3 + (dx + 1)] * ip[yy * WW + xx];
    }
  out[((size_t)bb * C + c) * P + p] = acc;
}

// Row L2 norms over pixels. grid=(192, NB), block=256
__global__ void rownorm(const float* __restrict__ in, float* __restrict__ norm,
                        int P, int CS, int coff) {
  int c = blockIdx.x, bb = blockIdx.y;
  const float* ip = in + ((size_t)bb * CS + coff + c) * P;
  __shared__ float red[256];
  float s = 0.f;
  for (int i = threadIdx.x; i < P; i += 256) { float v = ip[i]; s += v * v; }
  red[threadIdx.x] = s; __syncthreads();
  for (int st = 128; st > 0; st >>= 1) {
    if ((int)threadIdx.x < st) red[threadIdx.x] += red[threadIdx.x + st];
    __syncthreads();
  }
  if (threadIdx.x == 0) norm[bb * 192 + c] = fmaxf(sqrtf(red[0]), 1e-12f);
}

// ---------------------------------------------------------------------------
// Channel attention logits (32x32 per head, K=PIX reduction) + softmax.
// grid=(HEADS, NB), block=1024 (one thread per (c,e) pair)
// ---------------------------------------------------------------------------
__global__ void attn_logits(const float* __restrict__ Q, const float* __restrict__ KV,
                            const float* __restrict__ nq, const float* __restrict__ nk,
                            const float* __restrict__ temp, float* __restrict__ attn,
                            int P) {
  int h = blockIdx.x, bb = blockIdx.y;
  int e = threadIdx.x & 31, c = threadIdx.x >> 5;
  __shared__ float qs[32 * 64];
  __shared__ float ks[32 * 64];
  const float* qp = Q  + ((size_t)bb * DIMC + h * 32) * P;           // CS = 192
  const float* kp = KV + ((size_t)bb * (2 * DIMC) + h * 32) * P;     // CS = 384, k half
  float acc = 0.f;
  for (int j0 = 0; j0 < P; j0 += 64) {
    for (int i = threadIdx.x; i < 2048; i += 1024) {
      int row = i >> 6, col = i & 63;
      qs[i] = qp[(size_t)row * P + j0 + col];
      ks[i] = kp[(size_t)row * P + j0 + col];
    }
    __syncthreads();
#pragma unroll 8
    for (int j = 0; j < 64; ++j) acc += qs[c * 64 + j] * ks[e * 64 + j];
    __syncthreads();
  }
  acc = acc / (nq[bb * 192 + h * 32 + c] * nk[bb * 192 + h * 32 + e]) * temp[h];
  __shared__ float sl[32 * 32];
  __shared__ float rmax[32], rsum[32];
  sl[c * 32 + e] = acc; __syncthreads();
  if (threadIdx.x < 32) {
    int row = threadIdx.x;
    float mx = -1e30f;
    for (int j = 0; j < 32; ++j) mx = fmaxf(mx, sl[row * 32 + j]);
    float sm = 0.f;
    for (int j = 0; j < 32; ++j) sm += expf(sl[row * 32 + j] - mx);
    rmax[row] = mx; rsum[row] = sm;
  }
  __syncthreads();
  attn[(((size_t)bb * HEADS + h) * 32 + c) * 32 + e] = expf(sl[c * 32 + e] - rmax[c]) / rsum[c];
}

// out = attn @ v, written swizzled bf16 for the out-proj GEMM.
// grid=(P/64, HEADS, NB), block=256
__global__ void attn_apply(const float* __restrict__ attn, const float* __restrict__ KV,
                           bf16_t* __restrict__ O1, int P) {
  int h = blockIdx.y, bb = blockIdx.z;
  int p0 = blockIdx.x * 64;
  __shared__ float as[32 * 32];
  __shared__ float vs[32 * 64];
  for (int i = threadIdx.x; i < 1024; i += 256)
    as[i] = attn[(size_t)(bb * HEADS + h) * 1024 + i];
  const float* vp = KV + ((size_t)bb * (2 * DIMC) + DIMC + h * 32) * P;   // v half
  for (int i = threadIdx.x; i < 2048; i += 256) {
    int row = i >> 6, col = i & 63;
    vs[i] = vp[(size_t)row * P + p0 + col];
  }
  __syncthreads();
  int pl = threadIdx.x & 63;
  int cg = threadIdx.x >> 6;
#pragma unroll
  for (int j = 0; j < 8; ++j) {
    int c = cg * 8 + j;
    float acc = 0.f;
#pragma unroll 8
    for (int e = 0; e < 32; ++e) acc += as[c * 32 + e] * vs[e * 64 + pl];
    O1[(((size_t)bb * HEADS + h) * P + (p0 + pl)) * 32 + c] = f2bf(acc);
  }
}

// ---------------------------------------------------------------------------
// Fused depthwise 3x3 on [2,768,P] + exact GELU gate -> swizzled bf16 [384ch]
// grid=(P/256, 384, NB), block=256
// ---------------------------------------------------------------------------
__global__ void dw_gelu_gate(const float* __restrict__ Yp, const float* __restrict__ wt,
                             bf16_t* __restrict__ Yg, int P) {
  int p  = blockIdx.x * blockDim.x + threadIdx.x;
  int c  = blockIdx.y;
  int bb = blockIdx.z;
  if (p >= P) return;
  int py = p / WW, px = p % WW;
  const float* ia = Yp + ((size_t)bb * (2 * HIDDEN) + c) * P;
  const float* ig = Yp + ((size_t)bb * (2 * HIDDEN) + c + HIDDEN) * P;
  const float* wa = wt + c * 9;
  const float* wg = wt + (c + HIDDEN) * 9;
  float a = 0.f, g = 0.f;
#pragma unroll
  for (int dy = -1; dy <= 1; ++dy)
#pragma unroll
    for (int dx = -1; dx <= 1; ++dx) {
      int yy = py + dy, xx = px + dx;
      if (yy >= 0 && yy < HH && xx >= 0 && xx < WW) {
        int o = yy * WW + xx, wi = (dy + 1) * 3 + (dx + 1);
        a += wa[wi] * ia[o];
        g += wg[wi] * ig[o];
      }
    }
  float ge = 0.5f * a * (1.f + erff(a * 0.70710678118f));
  Yg[(((size_t)bb * 12 + (c >> 5)) * P + p) * 32 + (c & 31)] = f2bf(ge * g);
}

// copy k_v tail into d_out
__global__ void copy_kv(const float* __restrict__ src, float* __restrict__ dst, int n) {
  int i = blockIdx.x * blockDim.x + threadIdx.x;
  if (i < n) dst[i] = src[i];
}

// ---------------------------------------------------------------------------
// Workspace layout (bytes)
// ---------------------------------------------------------------------------
static const size_t O_SMALL = 0;                       // 64 KB of small vectors
static const size_t O_M1R   = 65536;                   // bf16 [2,160,P]  10.49 MB
static const size_t O_M1I   = O_M1R + 10485760;        // bf16 [2, 64,P]   4.19 MB
static const size_t O_WQR   = O_M1I + 4194304;         // frag bf16 192x160
static const size_t O_WKVI  = O_WQR + 61440;           // frag bf16 384x64
static const size_t O_WPO   = O_WKVI + 49152;          // frag bf16 192x192
static const size_t O_WPIN  = O_WPO + 73728;           // frag bf16 768x192
static const size_t O_WPOUT = O_WPIN + 294912;         // frag bf16 192x384
static const size_t O_QP    = O_WPOUT + 147456;        // f32 [2,192,P]   25.17 MB
static const size_t O_KVP   = O_QP + 25165824;         // f32 [2,384,P]   50.33 MB
static const size_t O_Q     = O_KVP + 50331648;        // f32 [2,192,P]   25.17 MB
static const size_t O_KVD   = O_Q + 25165824;          // f32 [2,384,P]   50.33 MB
static const size_t O_O1    = O_KVD + 50331648;        // bf16 [2,192,P]  12.58 MB
static const size_t O_X1    = O_O1 + 12582912;         // f32 [2,192,P]   25.17 MB
// aliases (sequential stream => safe):
static const size_t O_M2 = O_M1R;   // bf16 [2,192,P] fits in M1r+M1i region
static const size_t O_YP = O_QP;    // f32 [2,768,P] == Qp+KVp+Q region exactly
static const size_t O_YG = O_KVD;   // bf16 [2,384,P] fits in KVd region
// small-region float offsets
#define SV_KVR 0
#define SV_KVI 512
#define SV_KVF 1024
#define SV_NQ  2048
#define SV_NK  2560
#define SV_ATT 4096

extern "C" void kernel_launch(void* const* d_in, const int* in_sizes, int n_in,
                              void* d_out, int out_size, void* d_ws, size_t ws_size,
                              hipStream_t stream) {
  const float* x      = (const float*)d_in[0];
  const float* k_v    = (const float*)d_in[1];
  const float* ln1_w  = (const float*)d_in[2];
  const float* ln1_b  = (const float*)d_in[3];
  const float* temp   = (const float*)d_in[4];
  const float* w_kR   = (const float*)d_in[5];
  const float* w_kI   = (const float*)d_in[6];
  const float* w_qR   = (const float*)d_in[7];
  const float* w_qdw  = (const float*)d_in[8];
  const float* w_kvI  = (const float*)d_in[9];
  const float* w_kvdw = (const float*)d_in[10];
  const float* w_po   = (const float*)d_in[11];
  const float* ln2_w  = (const float*)d_in[12];
  const float* ln2_b  = (const float*)d_in[13];
  const float* w_ffk  = (const float*)d_in[14];
  const float* w_pin  = (const float*)d_in[15];
  const float* w_dw   = (const float*)d_in[16];
  const float* w_pout = (const float*)d_in[17];

  char* ws = (char*)d_ws;
  float*  small = (float*)(ws + O_SMALL);
  bf16_t* M1r   = (bf16_t*)(ws + O_M1R);
  bf16_t* M1i   = (bf16_t*)(ws + O_M1I);
  bf16_t* WqR   = (bf16_t*)(ws + O_WQR);
  bf16_t* WkvI  = (bf16_t*)(ws + O_WKVI);
  bf16_t* Wpo   = (bf16_t*)(ws + O_WPO);
  bf16_t* Wpin  = (bf16_t*)(ws + O_WPIN);
  bf16_t* Wpout = (bf16_t*)(ws + O_WPOUT);
  float*  Qp    = (float*)(ws + O_QP);
  float*  KVp   = (float*)(ws + O_KVP);
  float*  Q     = (float*)(ws + O_Q);
  float*  KVd   = (float*)(ws + O_KVD);
  bf16_t* O1    = (bf16_t*)(ws + O_O1);
  float*  X1    = (float*)(ws + O_X1);
  bf16_t* M2    = (bf16_t*)(ws + O_M2);
  float*  Yp    = (float*)(ws + O_YP);
  bf16_t* Yg    = (bf16_t*)(ws + O_YG);

  const int P = PIX;

  // --- weight prep (f32 -> bf16 WMMA fragments, K padded to mult of 32) ---
  wprep<<<(192 * 160 + 255) / 256, 256, 0, stream>>>(w_qR,   WqR,   192, 144, 160);
  wprep<<<(384 *  64 + 255) / 256, 256, 0, stream>>>(w_kvI,  WkvI,  384,  48,  64);
  wprep<<<(192 * 192 + 255) / 256, 256, 0, stream>>>(w_po,   Wpo,   192, 192, 192);
  wprep<<<(768 * 192 + 255) / 256, 256, 0, stream>>>(w_pin,  Wpin,  768, 192, 192);
  wprep<<<(192 * 384 + 255) / 256, 256, 0, stream>>>(w_pout, Wpout, 192, 384, 384);

  // --- k_v projections ---
  kvproj<<<3, 256, 0, stream>>>(k_v, w_kR, w_kI, w_ffk,
                                small + SV_KVR, small + SV_KVI, small + SV_KVF);

  // --- LN1 + modulate -> swizzled bf16 ---
  ln1_mod<<<dim3(P / 256, NB), 256, 0, stream>>>(x, ln1_w, ln1_b,
      small + SV_KVR, small + SV_KVI, M1r, M1i, P);

  // --- q / kv 1x1 convs via WMMA ---
  gemm_wmma<<<dim3(P / 128, 192 / 16, NB), 32, 0, stream>>>(WqR,  M1r, Qp,  nullptr, 192, 160, P);
  gemm_wmma<<<dim3(P / 128, 384 / 16, NB), 32, 0, stream>>>(WkvI, M1i, KVp, nullptr, 384,  64, P);

  // --- depthwise 3x3 ---
  dwconv3<<<dim3(P / 256, 192, NB), 256, 0, stream>>>(Qp,  w_qdw,  Q,   192, P);
  dwconv3<<<dim3(P / 256, 384, NB), 256, 0, stream>>>(KVp, w_kvdw, KVd, 384, P);

  // --- row L2 norms for q and k ---
  rownorm<<<dim3(192, NB), 256, 0, stream>>>(Q,   small + SV_NQ, P, 192, 0);
  rownorm<<<dim3(192, NB), 256, 0, stream>>>(KVd, small + SV_NK, P, 384, 0);

  // --- channel attention ---
  attn_logits<<<dim3(HEADS, NB), 1024, 0, stream>>>(Q, KVd,
      small + SV_NQ, small + SV_NK, temp, small + SV_ATT, P);
  attn_apply<<<dim3(P / 64, HEADS, NB), 256, 0, stream>>>(small + SV_ATT, KVd, O1, P);

  // --- out projection + residual (x) ---
  gemm_wmma<<<dim3(P / 128, 192 / 16, NB), 32, 0, stream>>>(Wpo, O1, X1, x, 192, 192, P);

  // --- LN2 + modulate ---
  ln2_mod<<<dim3(P / 256, NB), 256, 0, stream>>>(X1, ln2_w, ln2_b, small + SV_KVF, M2, P);

  // --- FFN in (768 ch) via WMMA ---
  gemm_wmma<<<dim3(P / 128, 768 / 16, NB), 32, 0, stream>>>(Wpin, M2, Yp, nullptr, 768, 192, P);

  // --- fused dwconv + GELU gate ---
  dw_gelu_gate<<<dim3(P / 256, HIDDEN, NB), 256, 0, stream>>>(Yp, w_dw, Yg, P);

  // --- FFN out + residual (X1) -> d_out ---
  gemm_wmma<<<dim3(P / 128, 192 / 16, NB), 32, 0, stream>>>(Wpout, Yg, (float*)d_out, X1, 192, 384, P);

  // --- k_v passthrough tail of d_out ---
  copy_kv<<<2, 256, 0, stream>>>(k_v, (float*)d_out + (size_t)NB * DIMC * PIX, NB * 256);
}
